// WTpNsynth_40965398069316
// MI455X (gfx1250) — matrix operand order repaired and checked by reading, not compiled, not run on hardware
//
#include <hip/hip_runtime.h>
#include <math.h>

#define SRATE   44100.0f
#define T_SAMP  262144
#define F_FRM   4096
#define NWT     20
#define WTL     512
#define NB      33
#define HLEN    44100
#define HPAD    45056            // 44 chunks of 1024
#define RCHUNK  1024
#define XW      (RCHUNK + 288)   // staged x window (elements), even
#define XDW     (XW / 2)         // 656 dwords
#define HW      (RCHUNK + 64)    // staged h window (elements)
#define HDW     (HW / 2)         // 544 dwords
#define HPKN    ((HPAD + 64) / 2) // 22560 dwords per packed-h array

typedef __attribute__((ext_vector_type(8)))  float        v8f;
typedef __attribute__((ext_vector_type(8)))  unsigned int v8u;
typedef __attribute__((ext_vector_type(16))) __bf16       v16bf;

__device__ __forceinline__ float exp_sig(float x) {
    float s = 1.0f / (1.0f + __expf(-x));
    return 2.0f * __powf(s, 2.3025850929940457f) + 1e-7f;
}

// jax.image.resize 'linear' upsample x64 with half-pixel centers + edge clamp
__device__ __forceinline__ float f0_up(const float* __restrict__ f0b, int t) {
    float c = (t + 0.5f) * (1.0f / 64.0f) - 0.5f;
    float fl = floorf(c);
    int i0 = (int)fl;
    float w = c - fl;
    int ia = i0 < 0 ? 0 : (i0 > F_FRM - 1 ? F_FRM - 1 : i0);
    int i1 = i0 + 1;
    int ib = i1 < 0 ? 0 : (i1 > F_FRM - 1 ? F_FRM - 1 : i1);
    return (1.0f - w) * f0b[ia] + w * f0b[ib];
}

__device__ __forceinline__ unsigned int bf16_rne(float x) {
    unsigned int u = __float_as_uint(x);
    return (u + 0x7FFFu + ((u >> 16) & 1u)) >> 16;
}
__device__ __forceinline__ float bf16_val(unsigned int b16) {
    return __uint_as_float(b16 << 16);
}
// split x ~= hi + lo, both representable in bf16
__device__ __forceinline__ void bf16_split(float x, unsigned int& hi, unsigned int& lo) {
    hi = bf16_rne(x);
    lo = bf16_rne(x - bf16_val(hi));
}

// ---------------- noise branch: IR synthesis + causal 64-tap conv ----------
__global__ __launch_bounds__(256) void k_noise(const float* __restrict__ nbands,
                                               const float* __restrict__ noise,
                                               float* __restrict__ out_noise) {
    __shared__ float simp[4][64];
    __shared__ float sns[4][64];
    __shared__ float snb[4][NB];
    int q = threadIdx.x >> 6;
    int j = threadIdx.x & 63;
    int frame = blockIdx.x * 4 + q;        // frame in [0, B*F)
    if (j < NB) snb[q][j] = exp_sig(nbands[frame * NB + j]);
    sns[q][j] = noise[frame * 64 + j];
    __syncthreads();
    // irfft(amp, n=64) at point j, then Hann fold from the roll/window/roll:
    float ph = 6.283185307179586f * (float)j * (1.0f / 64.0f);
    float acc = snb[q][0];
    #pragma unroll
    for (int k = 1; k < 32; k++) acc += 2.0f * snb[q][k] * __cosf(ph * (float)k);
    acc += snb[q][32] * __cosf(ph * 32.0f);
    float ir = acc * (1.0f / 64.0f);
    simp[q][j] = ir * (0.5f + 0.5f * __cosf(ph));
    __syncthreads();
    float o = 0.0f;
    for (int m = 0; m <= j; m++) o += sns[q][m] * simp[q][j - m];
    out_noise[frame * 64 + j] = o;         // (B,T) flat == frame*64+j
}

// ---------------- phase scan: chunk partial sums -----------------------------
__global__ __launch_bounds__(256) void k_scanA(const float* __restrict__ f0,
                                               float* __restrict__ chunkSum) {
    __shared__ float red[256];
    int b = blockIdx.x >> 7, c = blockIdx.x & 127;
    const float* f0b = f0 + b * F_FRM;
    int t0 = c * 2048 + threadIdx.x * 8;
    const float sc = 512.0f / 44100.0f;
    float s = 0.0f;
    #pragma unroll
    for (int i = 0; i < 8; i++) s += f0_up(f0b, t0 + i) * sc;
    red[threadIdx.x] = s;
    __syncthreads();
    for (int off = 128; off > 0; off >>= 1) {
        if (threadIdx.x < off) red[threadIdx.x] += red[threadIdx.x + off];
        __syncthreads();
    }
    if (threadIdx.x == 0) chunkSum[blockIdx.x] = red[0];
}

// ---------------- phase scan: per-batch exclusive scan of 128 chunks ---------
__global__ void k_scanB(const float* __restrict__ chunkSum,
                        float* __restrict__ chunkOff) {
    int b = threadIdx.x;
    if (b < 8) {
        float a = 0.0f;
        for (int c = 0; c < 128; c++) {
            chunkOff[b * 128 + c] = a;
            a += chunkSum[b * 128 + c];
        }
    }
}

// ---------------- reverb impulse -------------------------------------------
__global__ __launch_bounds__(256) void k_hprep(const float* __restrict__ rn,
                                               const float* __restrict__ decay,
                                               const float* __restrict__ wet,
                                               float* __restrict__ h) {
    int t = blockIdx.x * 256 + threadIdx.x;
    if (t >= HPAD) return;
    float v = 0.0f;
    if (t < HLEN) {
        float sp = log1pf(__expf(-decay[0]));
        float sw = 1.0f / (1.0f + __expf(-wet[0]));
        v = rn[t] * __expf(-sp * ((float)t / SRATE) * 500.0f) * sw;
        if (t == 0) v = 1.0f;
    }
    h[t] = v;
}

// ---------------- pack h into bf16 hi/lo, even + shifted alignment copies ----
// hpk layout: 4 arrays of HPKN dwords: [0]=hi_even [1]=lo_even [2]=hi_shift [3]=lo_shift
// dword d of even arrays packs h elements (2d-32, 2d-31); shift arrays (2d-31, 2d-30).
__global__ __launch_bounds__(256) void k_hpack(const float* __restrict__ h,
                                               unsigned int* __restrict__ hpk) {
    int d = blockIdx.x * 256 + threadIdx.x;
    if (d >= HPKN) return;
    int e0 = 2 * d - 32;
    float f0v = (e0     >= 0 && e0     < HPAD) ? h[e0]     : 0.0f;
    float f1v = (e0 + 1 >= 0 && e0 + 1 < HPAD) ? h[e0 + 1] : 0.0f;
    float f2v = (e0 + 2 >= 0 && e0 + 2 < HPAD) ? h[e0 + 2] : 0.0f;
    unsigned int h0, l0, h1, l1, h2, l2;
    bf16_split(f0v, h0, l0);
    bf16_split(f1v, h1, l1);
    bf16_split(f2v, h2, l2);
    hpk[0 * HPKN + d] = h0 | (h1 << 16);
    hpk[1 * HPKN + d] = l0 | (l1 << 16);
    hpk[2 * HPKN + d] = h1 | (h2 << 16);
    hpk[3 * HPKN + d] = l1 | (l2 << 16);
}

// ---------------- wavetable synth + dereverb --------------------------------
__global__ __launch_bounds__(256) void k_synth(const float* __restrict__ f0,
                                               const float* __restrict__ watt,
                                               const float* __restrict__ amp,
                                               const float* __restrict__ wts,
                                               const float* __restrict__ chunkOff,
                                               const float* __restrict__ out_noise,
                                               float* __restrict__ out_wt,
                                               float* __restrict__ out_der) {
    __shared__ float swt[NWT * WTL];      // 40 KB
    __shared__ float satt[32 * NWT];
    __shared__ float samp[32];
    __shared__ float sscan[256];
    int b = blockIdx.x >> 7, c = blockIdx.x & 127;
    int tid = threadIdx.x;

    for (int i = tid; i < NWT * WTL; i += 256) {
        float v = wts[i];
        if (i >= 4 * WTL) v = tanhf(v);
        swt[i] = v;
    }
    int fr0 = c * 32;                      // 32 frames per 2048-sample chunk
    for (int k = tid; k < 32; k += 256) {
        int fg = b * F_FRM + fr0 + k;
        float tmp[NWT], sum = 0.0f;
        #pragma unroll
        for (int n = 0; n < NWT; n++) { tmp[n] = exp_sig(watt[fg * NWT + n]); sum += tmp[n]; }
        float inv = 1.0f / sum;
        #pragma unroll
        for (int n = 0; n < NWT; n++) satt[k * NWT + n] = tmp[n] * inv;
        samp[k] = exp_sig(amp[fg]);
    }
    const float sc = 512.0f / 44100.0f;
    const float* f0b = f0 + b * F_FRM;
    int t0 = c * 2048 + tid * 8;
    float incs[8], run = 0.0f;
    #pragma unroll
    for (int i = 0; i < 8; i++) { run += f0_up(f0b, t0 + i) * sc; incs[i] = run; }
    sscan[tid] = run;
    __syncthreads();
    for (int off = 1; off < 256; off <<= 1) {
        float v = 0.0f;
        if (tid >= off) v = sscan[tid - off];
        __syncthreads();
        sscan[tid] += v;
        __syncthreads();
    }
    float inc0 = f0b[0] * sc;
    float base = chunkOff[b * 128 + c] + (sscan[tid] - run) - inc0;
    #pragma unroll
    for (int i = 0; i < 8; i++) {
        int t = t0 + i;
        float idx = fmodf(base + incs[i], 512.0f);
        if (idx < 0.0f) idx += 512.0f;
        int il = (int)idx;
        float al = idx - (float)il;
        int ih = (il + 1) & 511;
        int lf = (t >> 6) - fr0;
        float mix = 0.0f;
        #pragma unroll
        for (int n = 0; n < NWT; n++) {
            float lo = swt[n * WTL + il], hv = swt[n * WTL + ih];
            mix += (lo + al * (hv - lo)) * satt[lf * NWT + n];
        }
        float wsig = mix * samp[lf];
        int g = b * T_SAMP + t;
        out_wt[g] = wsig;
        out_der[g] = wsig + out_noise[g];
    }
}

// ---------------- reverb FIR: Toeplitz GEMM, bf16x3 split on -----------------
// v_wmma_f32_16x16x32_bf16.  D[i][j] = out[n0+16i+j];
// A[i][kk] = x[n0+16i+kk-kb2]; B[kk][j] = h[kb2+kk-j]; kb2 over multiples of 32
// covers every tap exactly once per output column.
// out = xh*hh + xl*hh + xh*hl  (bf16 split, ~2^-16 relative accuracy)
// h band staged via gfx1250 async DMA (global_load_async_to_lds_b128).
__global__ __launch_bounds__(256) void k_reverb(const float* __restrict__ der,
                                                const unsigned int* __restrict__ hpk,
                                                float* __restrict__ syn) {
    // bf16 data staged as packed dwords (element 2i in [15:0], 2i+1 in [31:16])
    __shared__ __align__(16) unsigned int xs_hi[8][XDW];   // 21 KB
    __shared__ __align__(16) unsigned int xs_lo[8][XDW];   // 21 KB
    __shared__ __align__(16) unsigned int hs[4][HDW];      // hi_e, lo_e, hi_s, lo_s

    int n0 = blockIdx.x * 256;
    int wv   = threadIdx.x >> 5;             // wave == batch
    int lane = threadIdx.x & 31;
    int m  = lane & 15, kh = lane >> 4;
    int p  = m & 1;                          // column parity selects h copy

    v8f acc = {};
    int kEnd = HPAD < (n0 + 256) ? HPAD : (n0 + 256);

    // per-lane constant index parts
    int adw = 8 * m + 8 * kh;                            // + (RCHUNK - c)/2  (+r, +4+r)
    int bdw = 8 * kh + ((32 - m - p) >> 1);              // + c/2  (+r)

    unsigned int hs_base = (unsigned int)(size_t)&hs[0][0];

    for (int kb = 0; kb < kEnd; kb += RCHUNK) {
        __syncthreads();
        // ---- async-DMA the packed h band [kb/2, kb/2+HDW) of all 4 arrays into LDS
        // 4 arrays * HDW/4 = 544 b128 transfers; ASYNCcnt-tracked
        for (int op = threadIdx.x; op < 4 * (HDW / 4); op += 256) {
            int arr = op / (HDW / 4);
            int i4  = (op - arr * (HDW / 4)) * 4;
            unsigned int ldsaddr = hs_base + (unsigned int)(arr * HDW + i4) * 4u;
            const unsigned int* g = hpk + arr * HPKN + (kb >> 1) + i4;
            asm volatile("global_load_async_to_lds_b128 %0, %1, off"
                         :: "v"(ldsaddr), "v"(g) : "memory");
        }
        // ---- stage x windows [n0-kb-RCHUNK, n0-kb+288) per batch as bf16 hi/lo
        for (int i = threadIdx.x; i < 8 * XDW; i += 256) {
            int bb = i / XDW, dw = i - bb * XDW;
            int g0 = n0 - kb - RCHUNK + 2 * dw;
            float x0 = (g0     >= 0 && g0     < T_SAMP) ? der[bb * T_SAMP + g0]     : 0.0f;
            float x1 = (g0 + 1 >= 0 && g0 + 1 < T_SAMP) ? der[bb * T_SAMP + g0 + 1] : 0.0f;
            unsigned int h0, l0, h1, l1;
            bf16_split(x0, h0, l0);
            bf16_split(x1, h1, l1);
            xs_hi[bb][dw] = h0 | (h1 << 16);
            xs_lo[bb][dw] = l0 | (l1 << 16);
        }
        // prefetch next chunk's x window into cache
        {
            int bb  = threadIdx.x >> 5;
            int off = (threadIdx.x & 31) * 40;
            int gpf = n0 - kb - 2 * RCHUNK + off;
            if (gpf >= 0 && gpf < T_SAMP) __builtin_prefetch(&der[bb * T_SAMP + gpf], 0, 0);
        }
        asm volatile("s_wait_asynccnt 0x0" ::: "memory");
        __syncthreads();

        const unsigned int* xh32 = &xs_hi[wv][0];
        const unsigned int* xl32 = &xs_lo[wv][0];
        const unsigned int* hh32 = p ? &hs[2][0] : &hs[0][0];
        const unsigned int* hl32 = p ? &hs[3][0] : &hs[1][0];

        for (int c = 0; c < RCHUNK; c += 32) {
            int ad = adw + ((RCHUNK - c) >> 1);
            int bd = bdw + (c >> 1);
            v8u ahu, alu, bhu, blu;
            #pragma unroll
            for (int r = 0; r < 4; r++) {
                ahu[r]     = xh32[ad + r];
                ahu[4 + r] = xh32[ad + 4 + r];
                alu[r]     = xl32[ad + r];
                alu[4 + r] = xl32[ad + 4 + r];
            }
            #pragma unroll
            for (int r = 0; r < 8; r++) {
                bhu[r] = hh32[bd + r];
                blu[r] = hl32[bd + r];
            }
            v16bf Ah = __builtin_bit_cast(v16bf, ahu);
            v16bf Al = __builtin_bit_cast(v16bf, alu);
            v16bf Bh = __builtin_bit_cast(v16bf, bhu);
            v16bf Bl = __builtin_bit_cast(v16bf, blu);
            acc = __builtin_amdgcn_wmma_f32_16x16x32_bf16(false, Ah, false, Bh, (short)0, acc, false, false);
            acc = __builtin_amdgcn_wmma_f32_16x16x32_bf16(false, Al, false, Bh, (short)0, acc, false, false);
            acc = __builtin_amdgcn_wmma_f32_16x16x32_bf16(false, Ah, false, Bl, (short)0, acc, false, false);
        }
    }
    // C/D layout: VGPR r -> row M = r (lanes 0-15) / r+8 (lanes 16-31), N = lane&15
    #pragma unroll
    for (int r = 0; r < 8; r++) {
        int n = n0 + 16 * (r + 8 * kh) + m;
        syn[wv * T_SAMP + n] = acc[r];
    }
}

extern "C" void kernel_launch(void* const* d_in, const int* in_sizes, int n_in,
                              void* d_out, int out_size, void* d_ws, size_t ws_size,
                              hipStream_t stream) {
    (void)in_sizes; (void)n_in; (void)out_size; (void)ws_size;
    const float* f0   = (const float*)d_in[0];
    const float* watt = (const float*)d_in[1];
    const float* nb   = (const float*)d_in[2];
    const float* amp  = (const float*)d_in[3];
    const float* noi  = (const float*)d_in[4];
    const float* wts  = (const float*)d_in[5];
    const float* rn   = (const float*)d_in[6];
    const float* dec  = (const float*)d_in[7];
    const float* wet  = (const float*)d_in[8];

    float* out   = (float*)d_out;
    float* o_syn = out;                    // synth
    float* o_der = out + 2097152;          // dereverb
    float* o_noi = out + 4194304;          // noise_audio
    float* o_wt  = out + 6291456;          // wavetable_signal

    float* ws = (float*)d_ws;
    float* h  = ws;                        // HPAD floats
    float* cs = ws + HPAD;                 // 1024 chunk sums
    float* co = cs + 1024;                 // 1024 chunk offsets
    unsigned int* hpk = (unsigned int*)(co + 1024);  // 4*HPKN dwords packed bf16 h

    k_noise<<<8192, 256, 0, stream>>>(nb, noi, o_noi);
    k_scanA<<<1024, 256, 0, stream>>>(f0, cs);
    k_scanB<<<1, 32, 0, stream>>>(cs, co);
    k_hprep<<<(HPAD + 255) / 256, 256, 0, stream>>>(rn, dec, wet, h);
    k_hpack<<<(HPKN + 255) / 256, 256, 0, stream>>>(h, hpk);
    k_synth<<<1024, 256, 0, stream>>>(f0, watt, amp, wts, co, o_noi, o_wt, o_der);
    k_reverb<<<1024, 256, 0, stream>>>(o_der, hpk, o_syn);
}
